// MemoryAugmentedNetwork_37864431681779
// MI455X (gfx1250) — matrix operand order repaired and checked by reading, not compiled, not run on hardware
//
#include <hip/hip_runtime.h>

// CDNA5 / gfx1250 WMMA types (probe-verified signatures)
typedef __attribute__((ext_vector_type(16))) __bf16 v16bf;
typedef __attribute__((ext_vector_type(8)))  float  v8f;
typedef unsigned short u16;
typedef unsigned int   u32;

#define N_EMBD     2048
#define NUM_SLOTS  32
#define ROWS_TOTAL 16384   // B*S = 4*4096

#if __has_builtin(__builtin_amdgcn_global_load_async_to_lds_b128) && \
    __has_builtin(__builtin_amdgcn_s_wait_asynccnt)
#define USE_ASYNC 1
#else
#define USE_ASYNC 0
#endif

__device__ __forceinline__ u16 f2bf(float f) {
    u32 u = __builtin_bit_cast(u32, f);
    u += 0x7fffu + ((u >> 16) & 1u);     // round-to-nearest-even
    return (u16)(u >> 16);
}

__device__ __forceinline__ v8f vzero8() {
    v8f z = {0.f, 0.f, 0.f, 0.f, 0.f, 0.f, 0.f, 0.f};
    return z;
}

// A-fragment (16x32 bf16, MxK). Lane l (m=l&15, g=l>>4) holds
// K = g*8..+7 (VGPR0-3) and K = 16+g*8..+7 (VGPR4-7): two 16B chunks.
__device__ __forceinline__ v16bf load_a(const u16* __restrict__ base, int stride,
                                        int row, int kb, int g) {
    union { uint4 u[2]; v16bf v; } f;
    const u16* p = base + (size_t)row * stride + kb + g * 8;
    f.u[0] = *(const uint4*)(p);
    f.u[1] = *(const uint4*)(p + 16);   // +16 elements = +32 bytes
    return f.v;
}

// B-fragment (32x16 bf16, KxN) from a [N][K] (B^T) array:
// lane l (n=l&15, g=l>>4) holds K = g*8..+7 and K = 16+g*8..+7 (two 16B chunks).
__device__ __forceinline__ v16bf load_b(const u16* __restrict__ base, int stride,
                                        int col, int kb, int g) {
    union { uint4 u[2]; v16bf v; } f;
    const u16* p = base + (size_t)col * stride + kb + g * 8;
    f.u[0] = *(const uint4*)(p);
    f.u[1] = *(const uint4*)(p + 16);   // +16 elements = +32 bytes
    return f.v;
}

// Same fragment shapes read out of an LDS panel row (row stride 40 bf16 = 80B)
__device__ __forceinline__ v16bf lds_frag_a(const u16* rowp, int g) {
    union { uint4 u[2]; v16bf v; } f;
    f.u[0] = *(const uint4*)(rowp + g * 8);
    f.u[1] = *(const uint4*)(rowp + 16 + g * 8);
    return f.v;
}
__device__ __forceinline__ v16bf lds_frag_b(const u16* rowp, int g) {
    union { uint4 u[2]; v16bf v; } f;
    f.u[0] = *(const uint4*)(rowp + g * 8);
    f.u[1] = *(const uint4*)(rowp + 16 + g * 8);
    return f.v;
}

__device__ __forceinline__ v8f wmma_bf16(v16bf a, v16bf b, v8f c) {
    return __builtin_amdgcn_wmma_f32_16x16x32_bf16(
        /*neg_a=*/false, a, /*neg_b=*/false, b,
        /*c_mod=*/(short)0, c, /*reuse_a=*/false, /*reuse_b=*/false);
}

#if USE_ASYNC
// Builtin expects pointers to 16-byte int vectors in AS1 (global) / AS3 (LDS)
typedef int i4v __attribute__((vector_size(16)));
typedef __attribute__((address_space(1))) i4v glb_i4v;
typedef __attribute__((address_space(3))) i4v lds_i4v;

// Cooperatively stage a 128-row x 32-K bf16 panel (8 KB) into LDS.
// 512 16B chunks; 128 threads x 4 iterations; 4 async b128 issues per wave
// per panel (ASYNCcnt-tracked, in-order retirement per wave).
__device__ __forceinline__ void async_panel(const u16* __restrict__ src, int base_row,
                                            int K, int kb, u16 (*buf)[40], int tid) {
    #pragma unroll
    for (int it = 0; it < 4; ++it) {
        int c   = it * 128 + tid;
        int row = c >> 2;
        int ch  = (c & 3) * 8;                 // element offset within the 32-K row
        const u16* gsrc = src + (size_t)(base_row + row) * K + kb + ch;
        __builtin_amdgcn_global_load_async_to_lds_b128(
            (glb_i4v*)(void*)gsrc, (lds_i4v*)(void*)&buf[row][ch],
            /*offset=*/0, /*cpol=*/0);
    }
}
#endif

// ---------------------------------------------------------------- conversions
__global__ void k_f32_to_bf16(const float* __restrict__ in, u16* __restrict__ out, int n) {
    int i = (blockIdx.x * blockDim.x + threadIdx.x) * 4;
    if (i < n) {
        float4 v = *(const float4*)(in + i);
        ushort4 o;
        o.x = f2bf(v.x); o.y = f2bf(v.y); o.z = f2bf(v.z); o.w = f2bf(v.w);
        *(ushort4*)(out + i) = o;
    }
}

// memT[d][m] = bf16(mem[m][d]); NUM_SLOTS*N_EMBD threads
__global__ void k_transpose_mem(const float* __restrict__ mem, u16* __restrict__ memT) {
    int i = blockIdx.x * blockDim.x + threadIdx.x;
    int d = i >> 5, m = i & 31;
    memT[i] = f2bf(mem[(size_t)m * N_EMBD + d]);
}

// ---------------------------------------------------------------- big GEMMs
// C[row,col] = sum_k A[row,k] * W[col,k] + bias[col] (+ X residual)
// Block: 128 threads = 4 waves (2x2), each wave a 64x64 tile (4x4 WMMA tiles).
// Async path: 128x32 A and B panels staged in LDS, double buffered.
template <bool RESIDUAL>
__global__ __launch_bounds__(128)
void k_gemm(const u16* __restrict__ A, const u16* __restrict__ Bw,
            const float* __restrict__ bias, const float* __restrict__ X,
            u16* __restrict__ OutBf, float* __restrict__ OutF) {
    const int K = N_EMBD, N = N_EMBD;
    const int lane = threadIdx.x & 31;
    const int wv   = threadIdx.x >> 5;
    const int g    = lane >> 4, ln = lane & 15;
    const int m0 = blockIdx.y * 128 + (wv >> 1) * 64;
    const int n0 = blockIdx.x * 128 + (wv & 1) * 64;

    v8f acc[4][4];
    #pragma unroll
    for (int mt = 0; mt < 4; ++mt)
        #pragma unroll
        for (int nt = 0; nt < 4; ++nt) acc[mt][nt] = vzero8();

#if USE_ASYNC
    __shared__ __align__(16) u16 Abuf[2][128][40];   // 20 KB (row stride 80B)
    __shared__ __align__(16) u16 Bbuf[2][128][40];   // 20 KB
    const int tid  = threadIdx.x;
    const int mblk = blockIdx.y * 128;
    const int nblk = blockIdx.x * 128;
    const int arow = (wv >> 1) * 64;   // wave's row base inside the A panel
    const int brow = (wv & 1) * 64;    // wave's col base inside the B panel
    const int KSTEPS = K / 32;

    async_panel(A,  mblk, K, 0, Abuf[0], tid);
    async_panel(Bw, nblk, K, 0, Bbuf[0], tid);

    for (int ks = 0; ks < KSTEPS; ++ks) {
        const int cur = ks & 1;
        if (ks + 1 < KSTEPS) {
            async_panel(A,  mblk, K, (ks + 1) * 32, Abuf[cur ^ 1], tid);
            async_panel(Bw, nblk, K, (ks + 1) * 32, Bbuf[cur ^ 1], tid);
            // 8 newer ops outstanding; in-order retirement => oldest buffer done
            __builtin_amdgcn_s_wait_asynccnt(8);
        } else {
            __builtin_amdgcn_s_wait_asynccnt(0);
        }
        __syncthreads();   // all waves' shares of buf[cur] visible

        v16bf a[4], b[4];
        #pragma unroll
        for (int mt = 0; mt < 4; ++mt)
            a[mt] = lds_frag_a(&Abuf[cur][arow + mt * 16 + ln][0], g);
        #pragma unroll
        for (int nt = 0; nt < 4; ++nt)
            b[nt] = lds_frag_b(&Bbuf[cur][brow + nt * 16 + ln][0], g);
        #pragma unroll
        for (int mt = 0; mt < 4; ++mt)
            #pragma unroll
            for (int nt = 0; nt < 4; ++nt)
                acc[mt][nt] = wmma_bf16(a[mt], b[nt], acc[mt][nt]);

        __syncthreads();   // done reading buf[cur] before it is re-filled
    }
#else
    for (int kb = 0; kb < K; kb += 32) {
        v16bf a[4], b[4];
        #pragma unroll
        for (int mt = 0; mt < 4; ++mt) a[mt] = load_a(A, K, m0 + mt * 16 + ln, kb, g);
        #pragma unroll
        for (int nt = 0; nt < 4; ++nt) b[nt] = load_b(Bw, K, n0 + nt * 16 + ln, kb, g);
        #pragma unroll
        for (int mt = 0; mt < 4; ++mt)
            #pragma unroll
            for (int nt = 0; nt < 4; ++nt)
                acc[mt][nt] = wmma_bf16(a[mt], b[nt], acc[mt][nt]);
    }
#endif

    // C/D layout: lane holds col = ln, rows v + 8*g across 8 VGPRs
    #pragma unroll
    for (int mt = 0; mt < 4; ++mt) {
        #pragma unroll
        for (int nt = 0; nt < 4; ++nt) {
            int col = n0 + nt * 16 + ln;
            float bv = bias[col];
            #pragma unroll
            for (int v = 0; v < 8; ++v) {
                int row = m0 + mt * 16 + v + 8 * g;
                size_t idx = (size_t)row * N + col;
                float val = acc[mt][nt][v] + bv;
                if (RESIDUAL) {
                    OutF[idx] = val + X[idx];
                } else {
                    OutBf[idx] = f2bf(val);
                }
            }
        }
    }
}

// ------------------------------------------- attn + softmax + w@mem (fused)
// Each wave owns 16 rows: attn[16,32] via WMMA over K=2048, softmax in-regs
// with half-wave shuffles, w restaged through LDS into A-layout, then 128
// single-WMMA output tiles (K=32) against memT.
__global__ __launch_bounds__(256)
void k_attn(const u16* __restrict__ Q, const u16* __restrict__ Mb,
            const u16* __restrict__ MTb, u16* __restrict__ Ob) {
    __shared__ __align__(16) u16 wlds[8][16][32];
    const int K = N_EMBD;
    const int lane = threadIdx.x & 31;
    const int wv   = threadIdx.x >> 5;
    const int g    = lane >> 4, ln = lane & 15;
    const int r0 = (blockIdx.x * 8 + wv) * 16;

    v8f c0 = vzero8(), c1 = vzero8();
    for (int kb = 0; kb < K; kb += 32) {
        v16bf a  = load_a(Q, K, r0 + ln, kb, g);
        v16bf b0 = load_b(Mb, K, ln,      kb, g);   // slots 0..15
        v16bf b1 = load_b(Mb, K, 16 + ln, kb, g);   // slots 16..31
        c0 = wmma_bf16(a, b0, c0);
        c1 = wmma_bf16(a, b1, c1);
    }

    const float scale = 0.022097086912079608f;  // 1/sqrt(2048)
    float w0[8], w1[8];
    #pragma unroll
    for (int v = 0; v < 8; ++v) {
        // row (v + 8g): its 32 logits live in 16 lanes of this half-wave x 2 tiles
        float x0 = c0[v] * scale, x1 = c1[v] * scale;
        float mx = fmaxf(x0, x1);
        #pragma unroll
        for (int o = 1; o < 16; o <<= 1) mx = fmaxf(mx, __shfl_xor(mx, o, 16));
        float e0 = __expf(x0 - mx), e1 = __expf(x1 - mx);
        float s = e0 + e1;
        #pragma unroll
        for (int o = 1; o < 16; o <<= 1) s += __shfl_xor(s, o, 16);
        float inv = 1.0f / s;
        w0[v] = e0 * inv;
        w1[v] = e1 * inv;
    }

    // stage w [16 rows x 32 slots] to LDS so we can reload in A-fragment layout
    #pragma unroll
    for (int v = 0; v < 8; ++v) {
        int row = v + 8 * g;
        wlds[wv][row][ln]      = f2bf(w0[v]);
        wlds[wv][row][16 + ln] = f2bf(w1[v]);
    }
    __syncthreads();

    union { uint4 u[2]; v16bf v; } af;
    const u16* wp = &wlds[wv][ln][0];
    af.u[0] = *(const uint4*)(wp + g * 8);        // K = g*8 .. +7
    af.u[1] = *(const uint4*)(wp + 16 + g * 8);   // K = 16+g*8 .. +7

    for (int nb = 0; nb < N_EMBD / 16; ++nb) {
        v16bf b = load_b(MTb, NUM_SLOTS, nb * 16 + ln, 0, g);  // memT[d][m], K=32
        v8f d = wmma_bf16(af.v, b, vzero8());
        #pragma unroll
        for (int v = 0; v < 8; ++v) {
            int row = r0 + v + 8 * g;
            Ob[(size_t)row * N_EMBD + nb * 16 + ln] = f2bf(d[v]);
        }
    }
}

// ---------------------------------------------------------------- launcher
extern "C" void kernel_launch(void* const* d_in, const int* in_sizes, int n_in,
                              void* d_out, int out_size, void* d_ws, size_t ws_size,
                              hipStream_t stream) {
    const float* x   = (const float*)d_in[0];
    const float* mem = (const float*)d_in[1];
    const float* Wq  = (const float*)d_in[2];
    const float* bq  = (const float*)d_in[3];
    const float* Wm  = (const float*)d_in[4];
    const float* bm  = (const float*)d_in[5];
    float* out = (float*)d_out;

    // workspace layout (bytes); total ~144.25 MB, region 0 reused for mem_out
    char* ws = (char*)d_ws;
    u16* xb   = (u16*)(ws + 0);                       //  64 MB  x (bf16), later mem_out
    u16* qb   = (u16*)(ws + (size_t)67108864);        //  64 MB  query (bf16)
    u16* wqb  = (u16*)(ws + (size_t)134217728);       //   8 MB  Wq (bf16)
    u16* wmb  = (u16*)(ws + (size_t)142606336);       //   8 MB  Wm (bf16)
    u16* memb = (u16*)(ws + (size_t)150994944);       // 128 KB  mem [32][2048] (bf16)
    u16* memT = (u16*)(ws + (size_t)151126016);       // 128 KB  memT [2048][32] (bf16)
    u16* ob   = xb;                                   // mem_out reuses x region

    const int nx = ROWS_TOTAL * N_EMBD;   // 33554432
    const int nw = N_EMBD * N_EMBD;       //  4194304
    const int nm = NUM_SLOTS * N_EMBD;    //    65536

    k_f32_to_bf16<<<nx / 4 / 256, 256, 0, stream>>>(x,   xb,  nx);
    k_f32_to_bf16<<<nw / 4 / 256, 256, 0, stream>>>(Wq,  wqb, nw);
    k_f32_to_bf16<<<nw / 4 / 256, 256, 0, stream>>>(Wm,  wmb, nw);
    k_f32_to_bf16<<<nm / 4 / 256, 256, 0, stream>>>(mem, memb, nm);
    k_transpose_mem<<<nm / 256, 256, 0, stream>>>(mem, memT);

    dim3 gg(N_EMBD / 128, ROWS_TOTAL / 128);  // (16, 128)
    k_gemm<false><<<gg, 128, 0, stream>>>(xb, wqb, bq, nullptr, qb, nullptr);
    k_attn<<<ROWS_TOTAL / 16 / 8, 256, 0, stream>>>(qb, memb, memT, ob);
    k_gemm<true><<<gg, 128, 0, stream>>>(ob, wmb, bm, x, nullptr, out);
}
